// MemoryNetwork_15564961481211
// MI455X (gfx1250) — matrix operand order repaired and checked by reading, not compiled, not run on hardware
//
#include <hip/hip_runtime.h>
#include <math.h>

// ---------------- problem constants ----------------
#define BQ      256        // queries
#define DIM     256        // feature dim
#define MROWS   500000     // memory rows
#define TOPK    32
#define MCHUNK  64         // memory rows staged per inner iteration
#define CHUNKS  31         // chunks per block
#define ROWS_PB (MCHUNK * CHUNKS)              // 1984
#define NB1     ((MROWS + ROWS_PB - 1) / ROWS_PB)   // 253 stage-1 blocks
#define QSTRIDE 132        // LDS dwords per qn row (528B: bank-spread, 16B aligned)
#define BSTRIDE 40         // LDS dwords per d-row in B staging (16B aligned, bank-spread)
#define SSTRIDE 66         // LDS floats per query row in sims buffer (bank-spread)
#define CAND_TOT (NB1 * TOPK)   // 8096 candidates per query
#define CAND_CAP 8192
#define NEG_INF (-__builtin_huge_valf())

// LDS layout offsets (bytes) for k_sims
#define OFF_QNL   0
#define OFF_BST   (OFF_QNL + 256 * QSTRIDE * 4)        // 135168
#define OFF_RNM   (OFF_BST + 256 * BSTRIDE * 4)        // 176128
#define OFF_SIM   (OFF_RNM + 256)                      // 176384
#define OFF_TLS   (OFF_SIM + 256 * SSTRIDE * 4)        // 243968
#define OFF_TLI   (OFF_TLS + BQ * TOPK * 4)            // 276736
#define SMEM1     (OFF_TLI + BQ * TOPK * 4)            // 309504  (<= 320KB/WG)

typedef __attribute__((ext_vector_type(16))) __bf16 v16bf;
typedef __attribute__((ext_vector_type(8)))  float  v8f;

union Frag16 { v16bf v; uint4 q[2]; };

static __device__ __forceinline__ unsigned short f2bf(float f) {
    unsigned int u = __float_as_uint(f);
    u += 0x7FFFu + ((u >> 16) & 1u);          // round-to-nearest-even
    return (unsigned short)(u >> 16);
}
static __device__ __forceinline__ float bf2f(unsigned short h) {
    return __uint_as_float(((unsigned int)h) << 16);
}

// ---------------- kernel 1: q = query@Wq^T + bq ; row-normalize ; emit bf16 ----
__global__ void k_project(const float* __restrict__ query,
                          const float* __restrict__ Wq,
                          const float* __restrict__ bq,
                          unsigned short* __restrict__ qn) {
    __shared__ float qrow[DIM];
    __shared__ float red[DIM];
    const int b = blockIdx.x, t = threadIdx.x;
    qrow[t] = query[b * DIM + t];
    __syncthreads();
    const float* w = Wq + t * DIM;            // row t of Wq (nn.Linear: q = x @ Wq^T)
    float acc = bq[t];
#pragma unroll 8
    for (int j = 0; j < DIM; ++j) acc += qrow[j] * w[j];
    red[t] = acc * acc;
    __syncthreads();
    for (int s = 128; s > 0; s >>= 1) {
        if (t < s) red[t] += red[t + s];
        __syncthreads();
    }
    const float inv = 1.0f / fmaxf(sqrtf(red[0]), 1e-8f);
    qn[b * DIM + t] = f2bf(acc * inv);
}

// ---------------- kernel 2: cosine scores via WMMA + per-block partial top-k ----
__global__ void __launch_bounds__(256, 1)
k_sims(const float* __restrict__ mem,
       const unsigned char* __restrict__ valid,
       const unsigned short* __restrict__ qn,
       float* __restrict__ cand_s,
       int*   __restrict__ cand_i) {
    extern __shared__ __align__(16) char smem[];
    unsigned int* qnl    = (unsigned int*)(smem + OFF_QNL);  // qn rows, stride 132 dwords
    unsigned int* bstage = (unsigned int*)(smem + OFF_BST);
    float* rnorm = (float*)(smem + OFF_RNM);
    float* sims  = (float*)(smem + OFF_SIM);
    float* tls   = (float*)(smem + OFF_TLS);
    int*   tli   = (int*)  (smem + OFF_TLI);

    const int t    = threadIdx.x;
    const int lane = t & 31;
    const int w    = t >> 5;                  // wave id 0..7
    const int hi   = lane >> 4;               // 0 for lanes 0-15, 1 for 16-31
    const int lm   = lane & 15;
    const int blockStart = blockIdx.x * ROWS_PB;

    // ---- stage qn into LDS once (row r -> qnl[r*QSTRIDE + 0..127], padded stride)
    {
        const unsigned int* qd = (const unsigned int*)qn;     // 32768 dwords
        for (int e = t; e < 256 * 128; e += 256) {
            const int r = e >> 7, cc = e & 127;
            qnl[r * QSTRIDE + cc] = qd[e];
        }
    }

    // init this thread's private top-k list (query t)
#pragma unroll
    for (int e = 0; e < TOPK; ++e) { tls[t * TOPK + e] = NEG_INF; tli[t * TOPK + e] = 0; }
    float curMin = NEG_INF; int minPos = 0;
    __syncthreads();

    for (int c = 0; c < CHUNKS; ++c) {
        const int m0 = blockStart + c * MCHUNK;

        // prefetch next chunk (speculative)
        {
            int pr = m0 + MCHUNK + (t >> 2);
            if (pr >= MROWS) pr = MROWS - 1;
            __builtin_prefetch(mem + (size_t)pr * DIM + (t & 3) * 64, 0, 1);
        }

        // ---- stage chunk: fp32 -> bf16, B-fragment layout: bstage[d][p] = {row 2p, row 2p+1} at col d
        {
            const int d = t;                  // 256 threads -> 256 columns
            for (int p = 0; p < MCHUNK / 2; ++p) {
                const int r0 = m0 + 2 * p;
                const float a = (r0     < MROWS) ? mem[(size_t)r0 * DIM + d]       : 0.0f;
                const float b = (r0 + 1 < MROWS) ? mem[(size_t)(r0 + 1) * DIM + d] : 0.0f;
                bstage[d * BSTRIDE + p] = ((unsigned int)f2bf(b) << 16) | (unsigned int)f2bf(a);
            }
        }
        __syncthreads();

        // ---- per-row inverse norms (from the bf16-staged data), wave-parallel
        for (int rl = 0; rl < 8; ++rl) {
            const int ml = w * 8 + rl;        // 0..63
            const int p = ml >> 1, half = ml & 1;
            float ss = 0.0f;
            for (int dd = lane; dd < DIM; dd += 32) {
                const unsigned int pk = bstage[dd * BSTRIDE + p];
                const float v = bf2f(half ? (unsigned short)(pk >> 16) : (unsigned short)(pk & 0xFFFF));
                ss += v * v;
            }
            for (int o = 16; o > 0; o >>= 1) ss += __shfl_xor(ss, o, 32);
            if (lane == 0) rnorm[ml] = 1.0f / fmaxf(sqrtf(ss), 1e-8f);
        }

        // ---- WMMA: this wave owns q-tiles {2w, 2w+1} x m-tiles {0..3}
        v8f acc[2][4];
#pragma unroll
        for (int i = 0; i < 2; ++i)
#pragma unroll
            for (int j = 0; j < 4; ++j)
#pragma unroll
                for (int r = 0; r < 8; ++r) acc[i][j][r] = 0.0f;

        for (int ks = 0; ks < 8; ++ks) {      // K = 8 * 32 = 256
            Frag16 a[2];
#pragma unroll
            for (int i = 0; i < 2; ++i) {
                const int row = (w * 2 + i) * 16 + lm;              // A: lane%16 = M
                // A fragment from LDS: dword col = ks*16 + hi*8 (lane/16 selects K half)
                const uint4* ap = (const uint4*)(qnl + row * QSTRIDE + ks * 16 + (hi << 3));
                a[i].q[0] = ap[0]; a[i].q[1] = ap[1];
            }
#pragma unroll
            for (int j = 0; j < 4; ++j) {
                Frag16 bf;                                           // B: lane%16 = N, element s = K (16*hi + s)
                const unsigned short* bs16 = (const unsigned short*)bstage;
                unsigned short* bh = (unsigned short*)&bf;
                const int ncol = j * 16 + lm;
#pragma unroll
                for (int s = 0; s < 16; ++s)
                    bh[s] = bs16[(ks * 32 + hi * 16 + s) * (BSTRIDE * 2) + ncol];
#pragma unroll
                for (int i = 0; i < 2; ++i)
                    acc[i][j] = __builtin_amdgcn_wmma_f32_16x16x32_bf16(
                        false, a[i].v, false, bf.v, (short)0, acc[i][j], false, false);
            }
        }
        __syncthreads();

        // ---- scale by 1/||m||, mask, scatter to sims LDS (C layout)
#pragma unroll
        for (int i = 0; i < 2; ++i) {
            const int qbase = (w * 2 + i) * 16 + (hi ? 8 : 0);
#pragma unroll
            for (int j = 0; j < 4; ++j) {
                const int col = j * 16 + lm;
                const int m = m0 + col;
                const float rn = rnorm[col];
                const bool ok = (m < MROWS) && (valid[m < MROWS ? m : 0] != 0);
#pragma unroll
                for (int r = 0; r < 8; ++r)
                    sims[(qbase + r) * SSTRIDE + col] = ok ? acc[i][j][r] * rn : NEG_INF;
            }
        }
        __syncthreads();

        // ---- thread t inserts chunk's 64 scores of query t into its top-32
        for (int cc = 0; cc < MCHUNK; ++cc) {
            const float s = sims[t * SSTRIDE + cc];
            if (s > curMin) {
                tls[t * TOPK + minPos] = s;
                tli[t * TOPK + minPos] = m0 + cc;
                curMin = tls[t * TOPK]; minPos = 0;
#pragma unroll
                for (int e = 1; e < TOPK; ++e) {
                    const float v = tls[t * TOPK + e];
                    if (v < curMin) { curMin = v; minPos = e; }
                }
            }
        }
        __syncthreads();
    }

    // ---- emit this block's candidates
    const size_t base = ((size_t)blockIdx.x * BQ + t) * TOPK;
#pragma unroll
    for (int e = 0; e < TOPK; ++e) {
        cand_s[base + e] = tls[t * TOPK + e];
        cand_i[base + e] = tli[t * TOPK + e];
    }
}

// ---------------- kernel 3: merge candidates -> global top-32 per query --------
__global__ void k_topmerge(const float* __restrict__ cand_s,
                           const int*   __restrict__ cand_i,
                           int* __restrict__ topidx) {
    extern __shared__ char sm2[];
    float* ls = (float*)sm2;                       // CAND_CAP floats
    int*   li = (int*)(sm2 + CAND_CAP * 4);        // CAND_CAP ints
    float* rs = (float*)(sm2 + CAND_CAP * 8);      // 256 floats
    int*   rp = (int*)(sm2 + CAND_CAP * 8 + 1024); // 256 ints

    const int q = blockIdx.x, t = threadIdx.x;
    for (int e = t; e < CAND_CAP; e += 256) {
        if (e < CAND_TOT) {
            const int blk = e >> 5, r = e & 31;
            ls[e] = cand_s[((size_t)blk * BQ + q) * TOPK + r];
            li[e] = cand_i[((size_t)blk * BQ + q) * TOPK + r];
        } else { ls[e] = NEG_INF; li[e] = 0x7FFFFFFF; }
    }
    __syncthreads();

    for (int it = 0; it < TOPK; ++it) {
        float bs = NEG_INF; int bp = -1;
        for (int e = t; e < CAND_CAP; e += 256) {
            const float s = ls[e];
            if (bp < 0 || s > bs || (s == bs && li[e] < li[bp])) { bs = s; bp = e; }
        }
        rs[t] = bs; rp[t] = bp;
        __syncthreads();
        for (int st = 128; st > 0; st >>= 1) {
            if (t < st) {
                const float s2 = rs[t + st]; const int p2 = rp[t + st];
                const bool take = (rp[t] < 0 && p2 >= 0) || (p2 >= 0 && s2 > rs[t]) ||
                                  (p2 >= 0 && rp[t] >= 0 && s2 == rs[t] && li[p2] < li[rp[t]]);
                if (take) { rs[t] = s2; rp[t] = p2; }
            }
            __syncthreads();
        }
        if (t == 0) {
            const int bestp = rp[0];
            topidx[q * TOPK + it] = (bestp >= 0) ? li[bestp] : 0;
            if (bestp >= 0) ls[bestp] = NEG_INF;
        }
        __syncthreads();
    }
}

// ---------------- kernel 4: gather retrieved rows, mean over queries -----------
__global__ void k_gather(const float* __restrict__ mem,
                         const int* __restrict__ topidx,
                         float* __restrict__ out) {
    __shared__ int idxs[BQ];
    const int t = threadIdx.x, kk = blockIdx.x;   // kk = rank 0..31
    idxs[t] = topidx[t * TOPK + kk];
    __syncthreads();
    float acc = 0.0f;
    for (int b = 0; b < BQ; ++b) acc += mem[(size_t)idxs[b] * DIM + t];
    out[kk * DIM + t] = acc * (1.0f / (float)BQ);
}

// ---------------- launcher -----------------------------------------------------
extern "C" void kernel_launch(void* const* d_in, const int* in_sizes, int n_in,
                              void* d_out, int out_size, void* d_ws, size_t ws_size,
                              hipStream_t stream) {
    const float* query          = (const float*)d_in[0];
    const float* memory         = (const float*)d_in[1];
    const unsigned char* valid  = (const unsigned char*)d_in[2];
    const float* Wq             = (const float*)d_in[3];
    const float* bq             = (const float*)d_in[4];
    // d_in[5] = k (=32, static in reference)

    char* ws = (char*)d_ws;
    unsigned short* qn = (unsigned short*)ws;                      // 256*256*2      = 131072
    size_t off = 131072;
    const size_t candBytes = (size_t)NB1 * BQ * TOPK * 4;          // 8,290,304
    float* cand_s = (float*)(ws + off);  off += candBytes;
    int*   cand_i = (int*)  (ws + off);  off += candBytes;
    int*   topidx = (int*)  (ws + off);                            // 256*32*4 = 32768

    const int smem2 = CAND_CAP * 8 + 256 * 8;                      // 67584

    k_project <<<BQ,  256, 0,     stream>>>(query, Wq, bq, qn);
    k_sims    <<<NB1, 256, SMEM1, stream>>>(memory, valid, qn, cand_s, cand_i);
    k_topmerge<<<BQ,  256, smem2, stream>>>(cand_s, cand_i, topidx);
    k_gather  <<<TOPK,256, 0,     stream>>>(memory, topidx, (float*)d_out);
}
